// GATEncoder_14542759264854
// MI455X (gfx1250) — compile-verified, matrix operand order checked by reading, not run on hardware
//
#include <hip/hip_runtime.h>
#include <hip/hip_bf16.h>
#include <math.h>

#define NUSR  50000
#define NFOOD 25000
#define NEDGE 500000
#define HDIM  128
#define NU_PAD 50000   // already multiple of 16
#define NF_PAD 25008   // 25000 padded to 16-row tiles

typedef __attribute__((ext_vector_type(16))) __bf16 v16bf;
typedef __attribute__((ext_vector_type(8)))  __bf16 v8bf;
typedef __attribute__((ext_vector_type(8)))  float  v8f;

// ---------------- conversions ----------------
__global__ void k_f32_to_bf16(const float* __restrict__ x, __bf16* __restrict__ y, int n) {
  int i = blockIdx.x * blockDim.x + threadIdx.x;
  if (i < n) y[i] = (__bf16)x[i];
}

// W [K,N] f32 -> Wt [N,K] bf16 (so the WMMA B fragment loads contiguously)
__global__ void k_transpose_bf16(const float* __restrict__ W, __bf16* __restrict__ Wt, int K, int N) {
  int i = blockIdx.x * blockDim.x + threadIdx.x;
  if (i >= K * N) return;
  int k = i / N, n = i - k * N;
  Wt[n * K + k] = (__bf16)W[i];
}

// ---------------- WMMA GEMM: C[Mpad,128] = A[Mpad,128] @ B[128,128] ----------------
// One wave computes a 16x64 strip: A fragments (4 k-steps) preloaded once into VGPRs,
// then 4 column tiles x 4 k-steps = 16 chained v_wmma_f32_16x16x32_bf16.
// Mpad is a multiple of 16 -> no guards, no branches in the epilogue.
__global__ void k_wmma_gemm128(const __bf16* __restrict__ A, const __bf16* __restrict__ Bt,
                               float* __restrict__ C, int Mtiles) {
  constexpr int K = 128;
  int wave = blockIdx.x * (blockDim.x >> 5) + (threadIdx.x >> 5);
  int lane = threadIdx.x & 31;
  int tm   = wave >> 1;                 // 16-row tile
  int half = wave & 1;                  // which 64-column half
  if (tm >= Mtiles) return;             // wave-uniform: EXEC all-ones for WMMA

  int lr    = lane & 15;
  int khalf = (lane >> 4) << 3;         // lanes 0-15: K 0..7/16..23 ; 16-31: 8..15/24..31
  const __bf16* arow = A + (size_t)(tm * 16 + lr) * K + khalf;

  v16bf afrag[4];
#pragma unroll
  for (int ks = 0; ks < 4; ks++) {
    v8bf lo = *(const v8bf*)(arow + ks * 32);
    v8bf hi = *(const v8bf*)(arow + ks * 32 + 16);
#pragma unroll
    for (int i = 0; i < 8; i++) { afrag[ks][i] = lo[i]; afrag[ks][i + 8] = hi[i]; }
  }

#pragma unroll
  for (int tn = 0; tn < 4; tn++) {
    const __bf16* brow = Bt + (size_t)(half * 64 + tn * 16 + lr) * K + khalf;
    v8f acc = {};
#pragma unroll
    for (int ks = 0; ks < 4; ks++) {
      v8bf lo = *(const v8bf*)(brow + ks * 32);
      v8bf hi = *(const v8bf*)(brow + ks * 32 + 16);
      v16bf bfrag;
#pragma unroll
      for (int i = 0; i < 8; i++) { bfrag[i] = lo[i]; bfrag[i + 8] = hi[i]; }
      acc = __builtin_amdgcn_wmma_f32_16x16x32_bf16(false, afrag[ks], false, bfrag,
                                                    (short)0, acc, false, false);
    }
    // C/D layout: VGPR i, lanes 0-15 -> M=i ; lanes 16-31 -> M=8+i ; N = lane&15
    float* cp = C + (size_t)(tm * 16 + ((lane >> 4) << 3)) * K + half * 64 + tn * 16 + lr;
#pragma unroll
    for (int i = 0; i < 8; i++) cp[i * K] = acc[i];
  }
}

// ---------------- per-node attention scalar: out[r] = dot(H[r,:], a) ----------------
__global__ void k_rowdot(const float* __restrict__ H, const float* __restrict__ a,
                         float* __restrict__ out, int Nrows) {
  int wave = blockIdx.x * (blockDim.x >> 5) + (threadIdx.x >> 5);
  int lane = threadIdx.x & 31;
  if (wave >= Nrows) return;
  float4 hv = ((const float4*)(H + (size_t)wave * HDIM))[lane];
  float4 av = ((const float4*)a)[lane];
  float s = hv.x * av.x + hv.y * av.y + hv.z * av.z + hv.w * av.w;
#pragma unroll
  for (int off = 16; off > 0; off >>= 1) s += __shfl_xor(s, off, 32);
  if (lane == 0) out[wave] = s;
}

// ---------------- edge passes ----------------
__device__ __forceinline__ unsigned f2mono(float x) {
  unsigned u = __float_as_uint(x);
  return (u & 0x80000000u) ? ~u : (u | 0x80000000u);
}
__device__ __forceinline__ float mono2f(unsigned t) {
  unsigned u = (t & 0x80000000u) ? (t ^ 0x80000000u) : ~t;
  return __uint_as_float(u);
}

__global__ void k_edge_score_max(const int* __restrict__ src, const int* __restrict__ dst,
                                 const float* __restrict__ asn, const float* __restrict__ adn,
                                 float* __restrict__ ebuf, unsigned* __restrict__ mmax, int E) {
  int i = blockIdx.x * blockDim.x + threadIdx.x;
  if (i >= E) return;
  float e = asn[src[i]] + adn[dst[i]];
  e = fmaxf(e, 0.2f * e);                       // leaky_relu(0.2)
  ebuf[i] = e;
  atomicMax(mmax + dst[i], f2mono(e));
}

__global__ void k_decode_max(const unsigned* __restrict__ mmax, float* __restrict__ m, int n) {
  int i = blockIdx.x * blockDim.x + threadIdx.x;
  if (i >= n) return;
  float v = mono2f(mmax[i]);                    // empty segment (0) decodes to NaN
  unsigned e = __float_as_uint(v) & 0x7F800000u;
  m[i] = (e != 0x7F800000u) ? v : 0.0f;         // where(isfinite(m), m, 0)
}

__global__ void k_edge_exp_den(const int* __restrict__ dst, const float* __restrict__ m,
                               float* __restrict__ ebuf, float* __restrict__ den, int E) {
  int i = blockIdx.x * blockDim.x + threadIdx.x;
  if (i >= E) return;
  float ee = __expf(ebuf[i] - m[dst[i]]);
  ebuf[i] = ee;
  atomicAdd(den + dst[i], ee);
}

// one wave32 per edge: coalesced float4 gather of hs[src], 4 float atomics/lane into out[dst]
__global__ void k_edge_aggregate(const int* __restrict__ src, const int* __restrict__ dst,
                                 const float* __restrict__ ebuf, const float* __restrict__ den,
                                 const float* __restrict__ HS, float* __restrict__ OUT, int E) {
  int g = blockIdx.x * blockDim.x + threadIdx.x;
  int edge = g >> 5, lane = g & 31;
  if (edge >= E) return;
  int s = src[edge], d = dst[edge];
  float alpha = ebuf[edge] / (den[d] + 1e-16f);
  float4 hv = ((const float4*)(HS + (size_t)s * HDIM))[lane];
  float* o = OUT + (size_t)d * HDIM + lane * 4;
  atomicAdd(o + 0, alpha * hv.x);
  atomicAdd(o + 1, alpha * hv.y);
  atomicAdd(o + 2, alpha * hv.z);
  atomicAdd(o + 3, alpha * hv.w);
}

// ---------------- BatchNorm (+ELU) ----------------
__global__ void k_bn_stats(const float* __restrict__ X, float* __restrict__ acc, int Nrows) {
  __shared__ float ssum[HDIM], ssq[HDIM];
  int tid = threadIdx.x;                                  // blockDim = 256
  if (tid < HDIM) { ssum[tid] = 0.f; ssq[tid] = 0.f; }
  __syncthreads();
  int rowsPerBlock = (Nrows + gridDim.x - 1) / gridDim.x;
  int r0 = blockIdx.x * rowsPerBlock;
  int r1 = min(r0 + rowsPerBlock, Nrows);
  int c = tid & 127, half = tid >> 7;
  float s = 0.f, q = 0.f;
  for (int r = r0 + half; r < r1; r += 2) {
    float v = X[(size_t)r * HDIM + c];
    s += v; q += v * v;
  }
  atomicAdd(&ssum[c], s); atomicAdd(&ssq[c], q);
  __syncthreads();
  if (tid < HDIM) { atomicAdd(acc + tid, ssum[tid]); atomicAdd(acc + HDIM + tid, ssq[tid]); }
}

__global__ void k_bn_apply(const float* __restrict__ X, const float* __restrict__ acc,
                           const float* __restrict__ g, const float* __restrict__ b,
                           float* __restrict__ outF, __bf16* __restrict__ outB, int Nrows) {
  int i = blockIdx.x * blockDim.x + threadIdx.x;
  if (i >= Nrows * HDIM) return;
  int c = i & 127;
  float inv = 1.0f / (float)Nrows;
  float mu  = acc[c] * inv;
  float var = acc[HDIM + c] * inv - mu * mu;               // biased var (torch BN normalize)
  float sc  = rsqrtf(var + 1e-5f) * g[c];
  float y = (X[i] - mu) * sc + b[c];
  y = y > 0.f ? y : (__expf(y) - 1.0f);                    // ELU(alpha=1)
  if (outF) outF[i] = y;
  if (outB) outB[i] = (__bf16)y;
}

// ---------------- health-preference MLP: tanh(LeakyReLU(x@hw1+hb1,0.01)@hw2+hb2) ----------------
__global__ void k_prefs(const float* __restrict__ XU2, const float* __restrict__ hw1,
                        const float* __restrict__ hb1, const float* __restrict__ hw2,
                        const float* __restrict__ hb2, float* __restrict__ prefs, int Nrows) {
  __shared__ float w1s[HDIM * 64];
  __shared__ float b1s[64], w2s[64];
  int tid = threadIdx.x;                                   // 256
  for (int i = tid; i < HDIM * 64; i += blockDim.x) w1s[i] = hw1[i];
  if (tid < 64) { b1s[tid] = hb1[tid]; w2s[tid] = hw2[tid]; }
  __syncthreads();
  int r = blockIdx.x * blockDim.x + tid;
  if (r >= Nrows) return;
  float hid[64];
#pragma unroll
  for (int j = 0; j < 64; j++) hid[j] = b1s[j];
  const float* row = XU2 + (size_t)r * HDIM;
  for (int k = 0; k < HDIM; k++) {
    float xv = row[k];
    const float* wr = &w1s[k * 64];
#pragma unroll
    for (int j = 0; j < 64; j++) hid[j] += xv * wr[j];
  }
  float p = hb2[0];
#pragma unroll
  for (int j = 0; j < 64; j++) {
    float h = hid[j];
    h = fmaxf(h, 0.01f * h);
    p += h * w2s[j];
  }
  prefs[r] = tanhf(p);
}

__global__ void k_health_scatter(const int* __restrict__ he_user, const int* __restrict__ he_food,
                                 const float* __restrict__ prefs, const float* __restrict__ scores,
                                 float* __restrict__ sfood, int E) {
  int i = blockIdx.x * blockDim.x + threadIdx.x;
  if (i >= E) return;
  atomicAdd(sfood + he_food[i], prefs[he_user[i]] * scores[i]);
}

__global__ void k_food_update(float* __restrict__ xf2, const float* __restrict__ sfood, int Nrows) {
  int i = blockIdx.x * blockDim.x + threadIdx.x;
  if (i >= Nrows * HDIM) return;
  xf2[i] += 0.1f * sfood[i >> 7];
}

// ---------------- host side ----------------
static inline int cdiv(int a, int b) { return (a + b - 1) / b; }

extern "C" void kernel_launch(void* const* d_in, const int* in_sizes, int n_in,
                              void* d_out, int out_size, void* d_ws, size_t ws_size,
                              hipStream_t stream) {
  (void)in_sizes; (void)n_in; (void)out_size; (void)ws_size;
  const float* x_user = (const float*)d_in[0];
  const float* x_food = (const float*)d_in[1];
  const int* uf_src = (const int*)d_in[2];
  const int* uf_dst = (const int*)d_in[3];
  const int* fu_src = (const int*)d_in[4];
  const int* fu_dst = (const int*)d_in[5];
  const int* he_user = (const int*)d_in[6];
  const int* he_food = (const int*)d_in[7];
  const float* scores = (const float*)d_in[8];
  // GAT params: per (L, et): w_src, w_dst, a_src, a_dst, b  -> indices 9..28
  const float* W_uf_src[2] = { (const float*)d_in[9],  (const float*)d_in[19] };
  const float* W_uf_dst[2] = { (const float*)d_in[10], (const float*)d_in[20] };
  const float* A_uf_src[2] = { (const float*)d_in[11], (const float*)d_in[21] };
  const float* A_uf_dst[2] = { (const float*)d_in[12], (const float*)d_in[22] };
  const float* W_fu_src[2] = { (const float*)d_in[14], (const float*)d_in[24] };
  const float* W_fu_dst[2] = { (const float*)d_in[15], (const float*)d_in[25] };
  const float* A_fu_src[2] = { (const float*)d_in[16], (const float*)d_in[26] };
  const float* A_fu_dst[2] = { (const float*)d_in[17], (const float*)d_in[27] };
  const float* BNg_u[2] = { (const float*)d_in[29], (const float*)d_in[33] };
  const float* BNb_u[2] = { (const float*)d_in[30], (const float*)d_in[34] };
  const float* BNg_f[2] = { (const float*)d_in[31], (const float*)d_in[35] };
  const float* BNb_f[2] = { (const float*)d_in[32], (const float*)d_in[36] };
  const float* hw1 = (const float*)d_in[37];
  const float* hb1 = (const float*)d_in[38];
  const float* hw2 = (const float*)d_in[39];
  const float* hb2 = (const float*)d_in[40];

  float* out_xu2   = (float*)d_out;
  float* out_xf2   = out_xu2 + (size_t)NUSR * HDIM;
  float* out_prefs = out_xf2 + (size_t)NFOOD * HDIM;

  // ---- workspace bump allocator (GEMM operands padded to 16-row tiles) ----
  char* wsb = (char*)d_ws;
  size_t off = 0;
  auto take = [&](size_t bytes) -> char* {
    char* p = wsb + off; off += (bytes + 255) & ~(size_t)255; return p;
  };
  __bf16* xu_bf = (__bf16*)take((size_t)NU_PAD * HDIM * 2);
  __bf16* xf_bf = (__bf16*)take((size_t)NF_PAD * HDIM * 2);
  float* hu_s = (float*)take((size_t)NU_PAD * HDIM * 4);
  float* hu_d = (float*)take((size_t)NU_PAD * HDIM * 4);
  float* hf_s = (float*)take((size_t)NF_PAD * HDIM * 4);
  float* hf_d = (float*)take((size_t)NF_PAD * HDIM * 4);
  float* au_s = (float*)take((size_t)NUSR * 4);
  float* au_d = (float*)take((size_t)NUSR * 4);
  float* af_s = (float*)take((size_t)NFOOD * 4);
  float* af_d = (float*)take((size_t)NFOOD * 4);
  unsigned* mm_u = (unsigned*)take((size_t)NUSR * 4);
  unsigned* mm_f = (unsigned*)take((size_t)NFOOD * 4);
  float* m_u = (float*)take((size_t)NUSR * 4);
  float* m_f = (float*)take((size_t)NFOOD * 4);
  float* den_u = (float*)take((size_t)NUSR * 4);
  float* den_f = (float*)take((size_t)NFOOD * 4);
  float* ebuf_uf = (float*)take((size_t)NEDGE * 4);
  float* ebuf_fu = (float*)take((size_t)NEDGE * 4);
  float* agg_u = (float*)take((size_t)NUSR * HDIM * 4);
  float* agg_f = (float*)take((size_t)NFOOD * HDIM * 4);
  float* bnacc_u = (float*)take(2 * HDIM * 4);
  float* bnacc_f = (float*)take(2 * HDIM * 4);
  __bf16* Wt[8];
  for (int i = 0; i < 8; i++) Wt[i] = (__bf16*)take((size_t)HDIM * HDIM * 2);
  float* sfood = (float*)take((size_t)NFOOD * 4);

  // ---- convert activations + weights to bf16 (weights transposed to [N,K]) ----
  k_f32_to_bf16<<<cdiv(NUSR * HDIM, 256), 256, 0, stream>>>(x_user, xu_bf, NUSR * HDIM);
  k_f32_to_bf16<<<cdiv(NFOOD * HDIM, 256), 256, 0, stream>>>(x_food, xf_bf, NFOOD * HDIM);
  const float* Wsrc[8] = { W_uf_src[0], W_uf_dst[0], W_fu_src[0], W_fu_dst[0],
                           W_uf_src[1], W_uf_dst[1], W_fu_src[1], W_fu_dst[1] };
  for (int i = 0; i < 8; i++)
    k_transpose_bf16<<<cdiv(HDIM * HDIM, 256), 256, 0, stream>>>(Wsrc[i], Wt[i], HDIM, HDIM);

  auto gemm = [&](const __bf16* A, const __bf16* Bt, float* C, int Mpad) {
    int Mtiles = Mpad / 16;
    int waves = Mtiles * 2;                     // 16x64 strip per wave
    k_wmma_gemm128<<<cdiv(waves, 8), 256, 0, stream>>>(A, Bt, C, Mtiles);
  };

  for (int L = 0; L < 2; L++) {
    // zero per-layer accumulators (graph-replay safe)
    hipMemsetAsync(mm_u, 0, (size_t)NUSR * 4, stream);     // 0 sorts below mono(-inf)
    hipMemsetAsync(mm_f, 0, (size_t)NFOOD * 4, stream);
    hipMemsetAsync(den_u, 0, (size_t)NUSR * 4, stream);
    hipMemsetAsync(den_f, 0, (size_t)NFOOD * 4, stream);
    hipMemsetAsync(agg_u, 0, (size_t)NUSR * HDIM * 4, stream);
    hipMemsetAsync(agg_f, 0, (size_t)NFOOD * HDIM * 4, stream);
    hipMemsetAsync(bnacc_u, 0, 2 * HDIM * 4, stream);
    hipMemsetAsync(bnacc_f, 0, 2 * HDIM * 4, stream);

    // projections (WMMA): user is uf-src & fu-dst; food is fu-src & uf-dst
    gemm(xu_bf, Wt[4 * L + 0], hu_s, NU_PAD);              // xu @ w_uf_src
    gemm(xu_bf, Wt[4 * L + 3], hu_d, NU_PAD);              // xu @ w_fu_dst
    gemm(xf_bf, Wt[4 * L + 2], hf_s, NF_PAD);              // xf @ w_fu_src
    gemm(xf_bf, Wt[4 * L + 1], hf_d, NF_PAD);              // xf @ w_uf_dst

    // attention scalars
    k_rowdot<<<cdiv(NUSR, 8), 256, 0, stream>>>(hu_s, A_uf_src[L], au_s, NUSR);
    k_rowdot<<<cdiv(NFOOD, 8), 256, 0, stream>>>(hf_d, A_uf_dst[L], af_d, NFOOD);
    k_rowdot<<<cdiv(NFOOD, 8), 256, 0, stream>>>(hf_s, A_fu_src[L], af_s, NFOOD);
    k_rowdot<<<cdiv(NUSR, 8), 256, 0, stream>>>(hu_d, A_fu_dst[L], au_d, NUSR);

    // uf edges: user -> food
    k_edge_score_max<<<cdiv(NEDGE, 256), 256, 0, stream>>>(uf_src, uf_dst, au_s, af_d, ebuf_uf, mm_f, NEDGE);
    k_decode_max<<<cdiv(NFOOD, 256), 256, 0, stream>>>(mm_f, m_f, NFOOD);
    k_edge_exp_den<<<cdiv(NEDGE, 256), 256, 0, stream>>>(uf_dst, m_f, ebuf_uf, den_f, NEDGE);
    k_edge_aggregate<<<cdiv(NEDGE * 32, 256), 256, 0, stream>>>(uf_src, uf_dst, ebuf_uf, den_f, hu_s, agg_f, NEDGE);
    // fu edges: food -> user
    k_edge_score_max<<<cdiv(NEDGE, 256), 256, 0, stream>>>(fu_src, fu_dst, af_s, au_d, ebuf_fu, mm_u, NEDGE);
    k_decode_max<<<cdiv(NUSR, 256), 256, 0, stream>>>(mm_u, m_u, NUSR);
    k_edge_exp_den<<<cdiv(NEDGE, 256), 256, 0, stream>>>(fu_dst, m_u, ebuf_fu, den_u, NEDGE);
    k_edge_aggregate<<<cdiv(NEDGE * 32, 256), 256, 0, stream>>>(fu_src, fu_dst, ebuf_fu, den_u, hf_s, agg_u, NEDGE);

    // BatchNorm + ELU (GAT bias cancels inside BN).  L1 -> bf16 recode, L2 -> d_out f32.
    k_bn_stats<<<256, 256, 0, stream>>>(agg_u, bnacc_u, NUSR);
    k_bn_stats<<<256, 256, 0, stream>>>(agg_f, bnacc_f, NFOOD);
    float*  oFu = (L == 0) ? nullptr : out_xu2;
    __bf16* oBu = (L == 0) ? xu_bf : nullptr;
    float*  oFf = (L == 0) ? nullptr : out_xf2;
    __bf16* oBf = (L == 0) ? xf_bf : nullptr;
    k_bn_apply<<<cdiv(NUSR * HDIM, 256), 256, 0, stream>>>(agg_u, bnacc_u, BNg_u[L], BNb_u[L], oFu, oBu, NUSR);
    k_bn_apply<<<cdiv(NFOOD * HDIM, 256), 256, 0, stream>>>(agg_f, bnacc_f, BNg_f[L], BNb_f[L], oFf, oBf, NFOOD);
  }

  // health-preference MLP + scatter_add update of xf2
  k_prefs<<<cdiv(NUSR, 256), 256, 0, stream>>>(out_xu2, hw1, hb1, hw2, hb2, out_prefs, NUSR);
  hipMemsetAsync(sfood, 0, (size_t)NFOOD * 4, stream);
  k_health_scatter<<<cdiv(NEDGE, 256), 256, 0, stream>>>(he_user, he_food, out_prefs, scores, sfood, NEDGE);
  k_food_update<<<cdiv(NFOOD * HDIM, 256), 256, 0, stream>>>(out_xf2, sfood, NFOOD);
}
